// TransformerBlock_5188320494353
// MI455X (gfx1250) — matrix operand order repaired and checked by reading, not compiled, not run on hardware
//
#include <hip/hip_runtime.h>

typedef _Float16 f16;
typedef _Float16 v8h  __attribute__((ext_vector_type(8)));
typedef _Float16 v16h __attribute__((ext_vector_type(16)));
typedef float    v8f  __attribute__((ext_vector_type(8)));

#define EMBED 1024
#define HEADS 16
#define HDIM  64
#define FFDIM 4096
#define BATCH 2
#define SEQ   2048
#define MTOT  (BATCH * SEQ)      // 4096 rows

// ---------------------------------------------------------------------------
// CDNA5 async global->LDS copy (ASYNCcnt-tracked, no VGPR round trip).
// ---------------------------------------------------------------------------
__device__ __forceinline__ void async_cp16(unsigned lds_off, const f16* g) {
  asm volatile("global_load_async_to_lds_b128 %0, %1, off"
               :: "v"(lds_off), "v"(g) : "memory");
}
#define WAIT_ASYNCCNT(N) asm volatile("s_wait_asynccnt %0" :: "n"(N) : "memory")

// ---------------------------------------------------------------------------
// VALU (DPP) max-reduce across each 16-lane half; progressively-uniform xor net
//   0xB1 quad_perm(1,0,3,2)=xor1, 0x4E quad_perm(2,3,0,1)=xor2,
//   0x141 row_half_mirror (=xor4 once quads uniform), 0x140 row_mirror (=xor8)
// ---------------------------------------------------------------------------
template <int CTRL>
__device__ __forceinline__ float maxdpp(float x) {
  int t = __builtin_amdgcn_update_dpp(0, __float_as_int(x), CTRL, 0xf, 0xf, true);
  return fmaxf(x, __int_as_float(t));
}
__device__ __forceinline__ float max16(float x) {
  x = maxdpp<0xB1>(x);
  x = maxdpp<0x4E>(x);
  x = maxdpp<0x141>(x);
  x = maxdpp<0x140>(x);
  return x;
}

// ---------------------------------------------------------------------------
// WMMA fragment builders (wave32, 16x16x32 f16)
// A layout: lane<16 -> K 0..15 contiguous, lane>=16 -> K 16..31
// B layout: lane<16 -> K 0..15 contiguous, lane>=16 -> K 16..31
// ---------------------------------------------------------------------------
__device__ __forceinline__ v16h make_fragA(const f16* p) {
  v8h lo = *(const v8h*)p;
  v8h hi = *(const v8h*)(p + 8);
  return __builtin_shufflevector(lo, hi, 0,1,2,3,4,5,6,7,8,9,10,11,12,13,14,15);
}
__device__ __forceinline__ v16h make_fragB(const f16* p) {
  v8h lo = *(const v8h*)p;
  v8h hi = *(const v8h*)(p + 8);
  return __builtin_shufflevector(lo, hi, 0,1,2,3,4,5,6,7,8,9,10,11,12,13,14,15);
}

// ---------------------------------------------------------------------------
// Weight pre-pass: f32 W[K][N] -> f16 Wt[N][K]   (tiled 32x32 transpose)
// ---------------------------------------------------------------------------
__global__ void __launch_bounds__(256)
wconv_kernel(const float* __restrict__ W, f16* __restrict__ Wt, int K, int N)
{
  __shared__ float tile[32][33];
  const int tx = threadIdx.x & 31;
  const int ty = threadIdx.x >> 5;
  const int n0 = blockIdx.x * 32;
  const int k0 = blockIdx.y * 32;
#pragma unroll
  for (int i = 0; i < 4; i++)
    tile[ty + i * 8][tx] = W[(size_t)(k0 + ty + i * 8) * N + n0 + tx];
  __syncthreads();
#pragma unroll
  for (int i = 0; i < 4; i++)
    Wt[(size_t)(n0 + ty + i * 8) * K + k0 + tx] = (f16)tile[tx][ty + i * 8];
}

// ---------------------------------------------------------------------------
// V pre-pass: qkv_h V slice [b,t][h,d] -> vt [b,h][d][t]  (f16 32x32 transpose)
// grid = (SEQ/32, HDIM/32, BATCH*HEADS)
// ---------------------------------------------------------------------------
__global__ void __launch_bounds__(256)
vtrans_kernel(const f16* __restrict__ qkv, f16* __restrict__ vt)
{
  __shared__ f16 tile[32][33];
  const int tx = threadIdx.x & 31;
  const int ty = threadIdx.x >> 5;
  const int t0 = blockIdx.x * 32;
  const int d0 = blockIdx.y * 32;
  const int bh = blockIdx.z;
  const int b  = bh >> 4, h = bh & 15;
#pragma unroll
  for (int i = 0; i < 4; i++)
    tile[ty + i * 8][tx] =
        qkv[(size_t)(b * SEQ + t0 + ty + i * 8) * 3072 + 2 * EMBED + h * HDIM + d0 + tx];
  __syncthreads();
#pragma unroll
  for (int i = 0; i < 4; i++)
    vt[((size_t)bh * HDIM + d0 + ty + i * 8) * SEQ + t0 + tx] = tile[tx][ty + i * 8];
}

// ---------------------------------------------------------------------------
// LayerNorm: one block (256 thr) per row of 1024, f32 in -> f16 out
// ---------------------------------------------------------------------------
__global__ void __launch_bounds__(256)
ln_kernel(const float* __restrict__ x, const float* __restrict__ g,
          const float* __restrict__ b, f16* __restrict__ out)
{
  __shared__ float red[8];
  const int row = blockIdx.x;
  const int tid = threadIdx.x;
  const float* xr = x + (size_t)row * EMBED;

  float v[4];
#pragma unroll
  for (int i = 0; i < 4; i++) v[i] = xr[tid + i * 256];

  float s = v[0] + v[1] + v[2] + v[3];
#pragma unroll
  for (int off = 16; off; off >>= 1) s += __shfl_xor(s, off, 32);
  if ((tid & 31) == 0) red[tid >> 5] = s;
  __syncthreads();
  float mu = 0.f;
#pragma unroll
  for (int i = 0; i < 8; i++) mu += red[i];
  mu *= (1.0f / EMBED);
  __syncthreads();

  float d2 = 0.f;
#pragma unroll
  for (int i = 0; i < 4; i++) { float d = v[i] - mu; d2 += d * d; }
#pragma unroll
  for (int off = 16; off; off >>= 1) d2 += __shfl_xor(d2, off, 32);
  if ((tid & 31) == 0) red[tid >> 5] = d2;
  __syncthreads();
  float var = 0.f;
#pragma unroll
  for (int i = 0; i < 8; i++) var += red[i];
  var *= (1.0f / EMBED);
  const float rstd = rsqrtf(var + 1e-5f);

#pragma unroll
  for (int i = 0; i < 4; i++) {
    const int c = tid + i * 256;
    out[(size_t)row * EMBED + c] = (f16)((v[i] - mu) * rstd * g[c] + b[c]);
  }
}

// ---------------------------------------------------------------------------
// GEMM: C[M,N] = A_f16[M,K] * Wt_f16[N,K]^T + bias
//  MODE 0: f16 out, Q cols (<1024) pre-scaled by 1/sqrt(HDIM)   (QKV proj)
//  MODE 1: f32 out + residual                                    (residual adds)
//  MODE 2: f16 gelu(out)                                         (FFN1)
// 256 thr, block tile 128x256, K-step 32, 8 waves x (64x64) = 16 WMMA/step,
// double-buffered LDS staged entirely with global_load_async_to_lds_b128.
// ---------------------------------------------------------------------------
#define BM 128
#define BN 256
#define BK 32
#define LDT 40   // padded LDS stride (halves); 80B rows -> conflict-free frags

template <int MODE>
__global__ void __launch_bounds__(256)
gemm_kernel(const f16* __restrict__ A, const f16* __restrict__ Wt,
            const float* __restrict__ bias, const float* __restrict__ res,
            f16* __restrict__ out_h, float* __restrict__ out_f,
            int M, int N, int K)
{
  __shared__ f16 As[2][BM * LDT];   // 2 x 10240 B
  __shared__ f16 Bs[2][BN * LDT];   // 2 x 20480 B

  const int tid  = threadIdx.x;
  const int lane = tid & 31;
  const int wid  = tid >> 5;
  const int wm   = wid & 1;            // 2 row groups of 64
  const int wn   = wid >> 1;           // 4 col groups of 64
  const int row0 = blockIdx.y * BM;
  const int col0 = blockIdx.x * BN;

  const int lrow = lane & 15;
  const int kbA  = (lane >> 4) << 4;   // 0 / 16
  const int kbB  = (lane >> 4) << 4;   // 0 / 16

  // staging: A = 128 rows x 32 halves (2 thr/row); B = 256 rows x 32 halves (1 thr/row)
  const int arow  = tid >> 1;
  const int akoff = (tid & 1) << 4;
  const f16* aglob = A  + (size_t)(row0 + arow) * K + akoff;
  const f16* bglob = Wt + (size_t)(col0 + tid) * K;

  auto issue = [&](int buf, int k0) {   // 6 async instructions per wave
    const unsigned la = (unsigned)(size_t)&As[buf][arow * LDT + akoff];
    async_cp16(la,      aglob + k0);
    async_cp16(la + 16, aglob + k0 + 8);
    const unsigned lb = (unsigned)(size_t)&Bs[buf][tid * LDT];
    const f16* bg = bglob + k0;
#pragma unroll
    for (int i = 0; i < 4; i++) async_cp16(lb + i * 16, bg + i * 8);
  };

  v8f zacc = {};
  v8f acc[4][4];
#pragma unroll
  for (int i = 0; i < 4; i++)
#pragma unroll
    for (int j = 0; j < 4; j++) acc[i][j] = zacc;

  issue(0, 0);
  int buf = 0;
  for (int k0 = 0; k0 < K; k0 += BK, buf ^= 1) {
    if (k0 + BK < K) {
      issue(buf ^ 1, k0 + BK);   // prefetch next tile into other buffer
      WAIT_ASYNCCNT(6);          // in-order: retires the 6 current-tile copies
    } else {
      WAIT_ASYNCCNT(0);
    }
    __syncthreads();

    v16h afrag[4], bfrag[4];
#pragma unroll
    for (int i = 0; i < 4; i++)
      afrag[i] = make_fragA(&As[buf][(wm * 64 + i * 16 + lrow) * LDT + kbA]);
#pragma unroll
    for (int j = 0; j < 4; j++)
      bfrag[j] = make_fragB(&Bs[buf][(wn * 64 + j * 16 + lrow) * LDT + kbB]);

#pragma unroll
    for (int i = 0; i < 4; i++)
#pragma unroll
      for (int j = 0; j < 4; j++)
        acc[i][j] = __builtin_amdgcn_wmma_f32_16x16x32_f16(
            false, afrag[i], false, bfrag[j], (short)0, acc[i][j], false, false);
    __syncthreads();
  }

  // Epilogue. C layout: lane -> N = lane&15, VGPR r -> M = r + (lane>=16)*8
#pragma unroll
  for (int j = 0; j < 4; j++) {
    const int gc = col0 + wn * 64 + j * 16 + lrow;
    const float bb = bias[gc];
    const float qscale = (MODE == 0 && gc < EMBED) ? 0.125f : 1.0f;  // fold 1/sqrt(64)
#pragma unroll
    for (int i = 0; i < 4; i++) {
      const int mbase = row0 + wm * 64 + i * 16 + ((lane >> 4) << 3);
#pragma unroll
      for (int r = 0; r < 8; r++) {
        const size_t idx = (size_t)(mbase + r) * N + gc;
        float v = acc[i][j][r] + bb;
        if (MODE == 0) {
          out_h[idx] = (f16)(v * qscale);
        } else if (MODE == 1) {
          out_f[idx] = v + res[idx];
        } else {
          v = 0.5f * v * (1.0f + erff(v * 0.70710678118654752f));
          out_h[idx] = (f16)v;
        }
      }
    }
  }
}

// ---------------------------------------------------------------------------
// Flash attention (non-causal), f16 in (Q pre-scaled), f16 out.
// grid = (B*HEADS, SEQ/64); 128 threads = 4 waves, each wave owns 16 Q rows.
// Q/K/V staged async (K/V double-buffered). Softmax row-sums ride the WMMA
// pipe (P x ones); row-max reduces on the VALU via DPP (no LDS port traffic).
// ---------------------------------------------------------------------------
#define ALD 72   // padded LDS stride in halves (144B)

__global__ void __launch_bounds__(128)
attn_kernel(const f16* __restrict__ qkv, const f16* __restrict__ vt,
            f16* __restrict__ out)
{
  __shared__ f16 Qs [64 * ALD];
  __shared__ f16 Ks [2][64 * ALD];
  __shared__ f16 Vts[2][64 * ALD];   // already transposed: [d][key]
  __shared__ f16 Ps [64 * ALD];      // per-wave 16x64 P tiles

  const int tid  = threadIdx.x;
  const int lane = tid & 31;
  const int wid  = tid >> 5;          // 0..3
  const int bh   = blockIdx.x;
  const int b    = bh >> 4;
  const int h    = bh & 15;
  const int q0   = blockIdx.y * 64;

  const int lrow  = lane & 15;
  const int kbA   = (lane >> 4) << 4;
  const int kbB   = (lane >> 4) << 4;
  const int mhalf = (lane >> 4) << 3;

  const int srow = tid >> 1;          // staging: 64 rows, 2 thr x 32 halves
  const int soff = (tid & 1) << 5;

  const f16* kglob = qkv + (size_t)(b * SEQ + srow) * 3072 + EMBED + h * HDIM + soff;
  const f16* vglob = vt + ((size_t)bh * HDIM + srow) * SEQ + soff;

  auto issueKV = [&](int buf, int kblk) {   // 8 async instructions per wave
    const f16* kp = kglob + (size_t)kblk * 64 * 3072;
    const unsigned lk = (unsigned)(size_t)&Ks[buf][srow * ALD + soff];
#pragma unroll
    for (int i = 0; i < 4; i++) async_cp16(lk + i * 16, kp + i * 8);
    const f16* vp = vglob + kblk * 64;
    const unsigned lv = (unsigned)(size_t)&Vts[buf][srow * ALD + soff];
#pragma unroll
    for (int i = 0; i < 4; i++) async_cp16(lv + i * 16, vp + i * 8);
  };

  // stage Q once (async, Q already scaled by 1/sqrt(HDIM))
  {
    const f16* qp = qkv + (size_t)(b * SEQ + q0 + srow) * 3072 + h * HDIM + soff;
    const unsigned lq = (unsigned)(size_t)&Qs[srow * ALD + soff];
#pragma unroll
    for (int i = 0; i < 4; i++) async_cp16(lq + i * 16, qp + i * 8);
  }
  issueKV(0, 0);

  // constant all-ones B fragment: osum = P x 1 -> row sums in every column
  v16h ones;
#pragma unroll
  for (int i = 0; i < 16; i++) ones[i] = (f16)1.0f;

  v8f zacc = {};
  float mrun[8];
  v8f oacc[4], osum;
#pragma unroll
  for (int r = 0; r < 8; r++) mrun[r] = -1e30f;
#pragma unroll
  for (int j = 0; j < 4; j++) oacc[j] = zacc;
  osum = zacc;

  const int NB = SEQ / 64;
  int buf = 0;
  for (int kblk = 0; kblk < NB; kblk++, buf ^= 1) {
    if (kblk + 1 < NB) {
      issueKV(buf ^ 1, kblk + 1);
      WAIT_ASYNCCNT(8);          // retires current K/V (and Q on iter 0)
    } else {
      WAIT_ASYNCCNT(0);
    }
    __syncthreads();

    // S = Q * K^T (16x64 per wave), head-dim = 64 (2 WMMA steps)
    v8f sacc[4];
#pragma unroll
    for (int j = 0; j < 4; j++) sacc[j] = zacc;
#pragma unroll
    for (int ks = 0; ks < 2; ks++) {
      v16h aq = make_fragA(&Qs[(wid * 16 + lrow) * ALD + ks * 32 + kbA]);
#pragma unroll
      for (int j = 0; j < 4; j++) {
        v16h bk = make_fragB(&Ks[buf][(j * 16 + lrow) * ALD + ks * 32 + kbB]);
        sacc[j] = __builtin_amdgcn_wmma_f32_16x16x32_f16(
            false, aq, false, bk, (short)0, sacc[j], false, false);
      }
    }

    // online softmax: row r lives in one 16-lane half, stats replicated.
    // Row-max reduces on the VALU (DPP); sums ride the WMMA pipe.
#pragma unroll
    for (int r = 0; r < 8; r++) {
      float m = fmaxf(fmaxf(sacc[0][r], sacc[1][r]), fmaxf(sacc[2][r], sacc[3][r]));
      m = max16(m);
      const float mnew = fmaxf(mrun[r], m);
      const float corr = __expf(mrun[r] - mnew);
      mrun[r] = mnew;
      osum[r] *= corr;
#pragma unroll
      for (int j = 0; j < 4; j++) oacc[j][r] *= corr;

      const int prow = (wid * 16 + mhalf + r) * ALD;
#pragma unroll
      for (int j = 0; j < 4; j++)
        Ps[prow + j * 16 + lrow] = (f16)__expf(sacc[j][r] - mnew);  // C-layout -> LDS
    }

    // O += P * V ; row-sum += P * 1   (K-dim = 64 keys, 2 WMMA steps)
#pragma unroll
    for (int ks = 0; ks < 2; ks++) {
      v16h ap = make_fragA(&Ps[(wid * 16 + lrow) * ALD + ks * 32 + kbA]);
#pragma unroll
      for (int j = 0; j < 4; j++) {
        v16h bv = make_fragB(&Vts[buf][(j * 16 + lrow) * ALD + ks * 32 + kbB]);
        oacc[j] = __builtin_amdgcn_wmma_f32_16x16x32_f16(
            false, ap, false, bv, (short)0, oacc[j], false, false);
      }
      osum = __builtin_amdgcn_wmma_f32_16x16x32_f16(
          false, ap, false, ones, (short)0, osum, false, false);
    }
    __syncthreads();
  }

  // normalize and write (B,T,H*D) f16
  float rinv[8];
#pragma unroll
  for (int r = 0; r < 8; r++) rinv[r] = 1.0f / osum[r];
#pragma unroll
  for (int j = 0; j < 4; j++) {
    const int col = h * HDIM + j * 16 + lrow;
#pragma unroll
    for (int r = 0; r < 8; r++) {
      const int t = q0 + wid * 16 + mhalf + r;
      out[(size_t)(b * SEQ + t) * EMBED + col] = (f16)(oacc[j][r] * rinv[r]);
    }
  }
}

// ---------------------------------------------------------------------------
// Launch sequence
// ---------------------------------------------------------------------------
extern "C" void kernel_launch(void* const* d_in, const int* in_sizes, int n_in,
                              void* d_out, int out_size, void* d_ws, size_t ws_size,
                              hipStream_t stream)
{
  const float* x     = (const float*)d_in[0];
  const float* qkv_w = (const float*)d_in[1];
  const float* qkv_b = (const float*)d_in[2];
  const float* out_w = (const float*)d_in[3];
  const float* out_b = (const float*)d_in[4];
  const float* ff_w1 = (const float*)d_in[5];
  const float* ff_b1 = (const float*)d_in[6];
  const float* ff_w2 = (const float*)d_in[7];
  const float* ff_b2 = (const float*)d_in[8];
  const float* ln1_g = (const float*)d_in[9];
  const float* ln1_b = (const float*)d_in[10];
  const float* ln2_g = (const float*)d_in[11];
  const float* ln2_b = (const float*)d_in[12];
  float* out = (float*)d_out;

  char* ws = (char*)d_ws;
  f16* xn_h    = (f16*)(ws);                      //  8 MB ln1(x) f16
  f16* qkv_h   = (f16*)(ws + ( 8ull << 20));      // 24 MB qkv f16 (Q pre-scaled)
  f16* attn_h  = (f16*)(ws + (32ull << 20));      //  8 MB attn out f16
  f16* xn2_h   = (f16*)(ws + (40ull << 20));      //  8 MB ln2(x1) f16
  f16* h_h     = (f16*)(ws + (48ull << 20));      // 32 MB gelu hidden f16
  f16* qkv_wt  = (f16*)(ws + (80ull << 20));      //  6 MB qkv_w^T f16
  f16* out_wt  = (f16*)(ws + (86ull << 20));      //  2 MB out_w^T f16
  f16* ff_w1t  = (f16*)(ws + (88ull << 20));      //  8 MB ff_w1^T f16
  f16* ff_w2t  = (f16*)(ws + (96ull << 20));      //  8 MB ff_w2^T f16
  f16* vt_h    = (f16*)(ws + (104ull << 20));     //  8 MB V^T f16  (total 112 MB)
  (void)in_sizes; (void)n_in; (void)out_size; (void)ws_size;

  // 0. one-shot weight convert+transpose to f16 [N][K]
  wconv_kernel<<<dim3(3 * EMBED / 32, EMBED / 32), 256, 0, stream>>>(qkv_w, qkv_wt, EMBED, 3 * EMBED);
  wconv_kernel<<<dim3(EMBED / 32,     EMBED / 32), 256, 0, stream>>>(out_w, out_wt, EMBED, EMBED);
  wconv_kernel<<<dim3(FFDIM / 32,     EMBED / 32), 256, 0, stream>>>(ff_w1, ff_w1t, EMBED, FFDIM);
  wconv_kernel<<<dim3(EMBED / 32,     FFDIM / 32), 256, 0, stream>>>(ff_w2, ff_w2t, FFDIM, EMBED);

  // 1. xn1 = LN1(x)
  ln_kernel<<<MTOT, 256, 0, stream>>>(x, ln1_g, ln1_b, xn_h);
  // 2. qkv = xn1 @ qkv_w + b (Q cols scaled by 1/8)   (M=4096, N=3072, K=1024)
  gemm_kernel<0><<<dim3(3 * EMBED / BN, MTOT / BM), 256, 0, stream>>>(
      xn_h, qkv_wt, qkv_b, nullptr, qkv_h, nullptr, MTOT, 3 * EMBED, EMBED);
  // 2b. transpose V per (b,h) for async B-layout staging
  vtrans_kernel<<<dim3(SEQ / 32, HDIM / 32, BATCH * HEADS), 256, 0, stream>>>(qkv_h, vt_h);
  // 3. attn = flash_attention(q,k,v)
  attn_kernel<<<dim3(BATCH * HEADS, SEQ / 64), 128, 0, stream>>>(qkv_h, vt_h, attn_h);
  // 4. x1 = x + attn @ out_w + b       -> d_out (f32)
  gemm_kernel<1><<<dim3(EMBED / BN, MTOT / BM), 256, 0, stream>>>(
      attn_h, out_wt, out_b, x, nullptr, out, MTOT, EMBED, EMBED);
  // 5. xn2 = LN2(x1)
  ln_kernel<<<MTOT, 256, 0, stream>>>(out, ln2_g, ln2_b, xn2_h);
  // 6. h = gelu(xn2 @ ff_w1 + b)       (N=4096)
  gemm_kernel<2><<<dim3(FFDIM / BN, MTOT / BM), 256, 0, stream>>>(
      xn2_h, ff_w1t, ff_b1, nullptr, h_h, nullptr, MTOT, FFDIM, EMBED);
  // 7. out = x1 + h @ ff_w2 + b        (K=4096; in-place on d_out, one touch/elem)
  gemm_kernel<1><<<dim3(EMBED / BN, MTOT / BM), 256, 0, stream>>>(
      h_h, ff_w2t, ff_b2, out, nullptr, out, MTOT, EMBED, FFDIM);
}
